// Mamba3DecoderBlock_2499670966926
// MI455X (gfx1250) — matrix operand order repaired and checked by reading, not compiled, not run on hardware
//
#include <hip/hip_runtime.h>
#include <hip/hip_bf16.h>

typedef __bf16 bf16;
typedef bf16  bf16x16 __attribute__((ext_vector_type(16)));
typedef float v8f     __attribute__((ext_vector_type(8)));

union BFrag {
    bf16x16 v;
    float4  f4[2];
    bf16    h[16];
};

__device__ inline bf16 f2bf(float f) {
    union { float f; unsigned u; } in; in.f = f;
    unsigned r = in.u + 0x7FFFu + ((in.u >> 16) & 1u);
    union { unsigned short s; bf16 b; } out; out.s = (unsigned short)(r >> 16);
    return out.b;
}
__device__ inline float bf2f(bf16 b) {
    union { unsigned short s; bf16 b; } in; in.b = b;
    union { unsigned u; float f; } out; out.u = ((unsigned)in.s) << 16;
    return out.f;
}
__device__ inline v8f wmma_bf16(bf16x16 a, bf16x16 b, v8f c) {
    return __builtin_amdgcn_wmma_f32_16x16x32_bf16(false, a, false, b, (short)0, c, false, false);
}

// CDNA5 async global->LDS copy (ASYNCcnt-tracked), 16B per lane.
__device__ inline void async_load_lds_b128(void* lds_dst, const void* gsrc) {
    unsigned lds = (unsigned)(uintptr_t)lds_dst;   // low 32 bits = LDS offset
    asm volatile("global_load_async_to_lds_b128 %0, %1, off"
                 :: "v"(lds), "v"(gsrc) : "memory");
}
__device__ inline void wait_asynccnt0() {
    asm volatile("s_wait_asynccnt 0x0" ::: "memory");
}

// ---------------------------------------------------------------------------
// fp32 -> bf16 convert
// ---------------------------------------------------------------------------
__global__ void cvt_bf16_kernel(const float* __restrict__ src, bf16* __restrict__ dst, int n) {
    int i = blockIdx.x * blockDim.x + threadIdx.x;
    if (i < n) dst[i] = f2bf(src[i]);
}

// ---------------------------------------------------------------------------
// LayerNorm over D=512, optional mask multiply, bf16 output
// ---------------------------------------------------------------------------
__global__ __launch_bounds__(256) void ln_kernel(const float* __restrict__ x,
                                                 const float* __restrict__ g,
                                                 const float* __restrict__ b,
                                                 const float* __restrict__ mask,
                                                 bf16* __restrict__ out) {
    const int D = 512;
    int row = blockIdx.x, tid = threadIdx.x;
    const float* xr = x + (size_t)row * D;
    float v0 = xr[tid], v1 = xr[tid + 256];
    __shared__ float red[256];
    red[tid] = v0 + v1; __syncthreads();
    for (int o = 128; o > 0; o >>= 1) { if (tid < o) red[tid] += red[tid + o]; __syncthreads(); }
    float mean = red[0] / D; __syncthreads();
    float d0 = v0 - mean, d1 = v1 - mean;
    red[tid] = d0 * d0 + d1 * d1; __syncthreads();
    for (int o = 128; o > 0; o >>= 1) { if (tid < o) red[tid] += red[tid + o]; __syncthreads(); }
    float inv = rsqrtf(red[0] / D + 1e-5f);
    float mv = mask ? mask[row] : 1.f;
    out[(size_t)row * D + tid]       = f2bf((d0 * inv * g[tid]       + b[tid])       * mv);
    out[(size_t)row * D + tid + 256] = f2bf((d1 * inv * g[tid + 256] + b[tid + 256]) * mv);
}

// ---------------------------------------------------------------------------
// Generic bf16 WMMA GEMM: out = A[MxK] @ W[NxK]^T (+bias)(+resid)
// Block tile 64x128, 8 waves -> 32x32 each. K multiple of 32, M multiple of 64.
// Double-buffered LDS; A tile staged with global_load_async_to_lds_b128.
// ---------------------------------------------------------------------------
__global__ __launch_bounds__(256) void gemm_bf16_kernel(
    const bf16* __restrict__ A, const bf16* __restrict__ W,
    const float* __restrict__ bias, const float* __restrict__ resid,
    float* __restrict__ outF, bf16* __restrict__ outB,
    int M, int N, int K) {
    __shared__ __attribute__((aligned(16))) bf16 As[2][64 * 40];   // [m][k] stride 40
    __shared__ __attribute__((aligned(16))) bf16 Bs[2][32 * 136];  // [k][n] stride 136
    int tid = threadIdx.x;
    int lane = tid & 31, wid = tid >> 5;
    int wm = wid >> 2, wn = wid & 3;
    int m0 = blockIdx.y * 64;
    int n0 = blockIdx.x * 128;

    v8f acc[2][2];
    #pragma unroll
    for (int i = 0; i < 2; i++)
        #pragma unroll
        for (int j = 0; j < 2; j++)
            acc[i][j] = (v8f){0.f, 0.f, 0.f, 0.f, 0.f, 0.f, 0.f, 0.f};

    int arow = tid >> 2, akg = (tid & 3) * 8;
    int base0 = (lane < 16) ? 0 : 8;
    const bf16* arowp = A + (size_t)(m0 + arow) * K + akg;

    int bn[2], bkg[2];
    #pragma unroll
    for (int i = 0; i < 2; i++) {
        int c = tid + i * 256;
        bn[i] = c >> 2;
        bkg[i] = (c & 3) * 8;
    }

    BFrag wreg[2];
    int nk = K / 32;

    // ---- prologue: stage tile 0 ----
    async_load_lds_b128(&As[0][arow * 40 + akg], arowp);
    #pragma unroll
    for (int i = 0; i < 2; i++) {
        if (n0 + bn[i] < N) wreg[i].f4[0] = *(const float4*)(W + (size_t)(n0 + bn[i]) * K + bkg[i]);
        else                wreg[i].f4[0] = (float4){0.f, 0.f, 0.f, 0.f};
        #pragma unroll
        for (int j = 0; j < 8; j++) Bs[0][(bkg[i] + j) * 136 + bn[i]] = wreg[i].h[j];
    }
    wait_asynccnt0();
    __syncthreads();

    for (int kt = 0; kt < nk; kt++) {
        int cur = kt & 1;
        int k1 = (kt + 1) * 32;
        bool more = (kt + 1) < nk;

        // issue next tile's async A copy + B global loads before the math
        if (more) {
            async_load_lds_b128(&As[1 - cur][arow * 40 + akg], arowp + k1);
            #pragma unroll
            for (int i = 0; i < 2; i++) {
                if (n0 + bn[i] < N) wreg[i].f4[0] = *(const float4*)(W + (size_t)(n0 + bn[i]) * K + k1 + bkg[i]);
                else                wreg[i].f4[0] = (float4){0.f, 0.f, 0.f, 0.f};
            }
            __builtin_prefetch(arowp + k1 + 32, 0, 1);
        }

        // fragments from current buffers
        BFrag af[2], bf[2];
        #pragma unroll
        for (int i = 0; i < 2; i++) {
            const bf16* ap = &As[cur][(wm * 32 + i * 16 + (lane & 15)) * 40];
            af[i].f4[0] = *(const float4*)(ap + base0);
            af[i].f4[1] = *(const float4*)(ap + base0 + 16);
        }
        #pragma unroll
        for (int j = 0; j < 2; j++) {
            const bf16* bp = &Bs[cur][wn * 32 + j * 16 + (lane & 15)];
            #pragma unroll
            for (int s = 0; s < 8; s++) {
                bf[j].h[s]     = bp[(base0 + s) * 136];
                bf[j].h[s + 8] = bp[(base0 + 16 + s) * 136];
            }
        }
        #pragma unroll
        for (int i = 0; i < 2; i++)
            #pragma unroll
            for (int j = 0; j < 2; j++)
                acc[i][j] = wmma_bf16(af[i].v, bf[j].v, acc[i][j]);

        // store next B tile (transposed) into the other buffer, wait DMA, barrier
        if (more) {
            #pragma unroll
            for (int i = 0; i < 2; i++)
                #pragma unroll
                for (int j = 0; j < 8; j++)
                    Bs[1 - cur][(bkg[i] + j) * 136 + bn[i]] = wreg[i].h[j];
            wait_asynccnt0();
        }
        __syncthreads();
    }

    int mrow = (lane < 16) ? 0 : 8;
    #pragma unroll
    for (int i = 0; i < 2; i++)
        #pragma unroll
        for (int j = 0; j < 2; j++)
            #pragma unroll
            for (int r = 0; r < 8; r++) {
                int gm = m0 + wm * 32 + i * 16 + mrow + r;
                int gn = n0 + wn * 32 + j * 16 + (lane & 15);
                if (gn < N) {
                    float vv = acc[i][j][r];
                    if (bias)  vv += bias[gn];
                    if (resid) vv += resid[(size_t)gm * N + gn];
                    if (outF)  outF[(size_t)gm * N + gn] = vv;
                    if (outB)  outB[(size_t)gm * N + gn] = f2bf(vv);
                }
            }
}

// ---------------------------------------------------------------------------
// Causal depthwise conv (K=4) + SiLU over xBC channels
// ---------------------------------------------------------------------------
__global__ void conv_silu_kernel(const float* __restrict__ zx, const float* __restrict__ cw,
                                 const float* __restrict__ cb, float* __restrict__ xBC,
                                 int S, int total) {
    int idx = blockIdx.x * blockDim.x + threadIdx.x;
    if (idx >= total) return;
    int c = idx % 1152;
    int row = idx / 1152;
    int s = row % S, b = row / S;
    float acc = cb[c];
    #pragma unroll
    for (int j = 0; j < 4; j++) {
        int t = s - 3 + j;
        if (t >= 0) acc += cw[c * 4 + j] * zx[(size_t)(b * S + t) * 2184 + 1024 + c];
    }
    xBC[(size_t)row * 1152 + c] = acc / (1.f + __expf(-acc));
}

// ---------------------------------------------------------------------------
// dt = softplus(dt_raw + bias); dexp = exp(dt * (-exp(A_log)))
// ---------------------------------------------------------------------------
__global__ void dt_kernel(const float* __restrict__ zx, const float* __restrict__ dt_bias,
                          const float* __restrict__ A_log, float* __restrict__ dt,
                          float* __restrict__ dexp, int total) {
    int idx = blockIdx.x * blockDim.x + threadIdx.x;
    if (idx >= total) return;
    int h = idx & 7, row = idx >> 3;
    float xv = zx[(size_t)row * 2184 + 2176 + h] + dt_bias[h];
    float sp = xv > 20.f ? xv : log1pf(__expf(xv));
    float Ah = -__expf(A_log[h]);
    dt[idx] = sp;
    dexp[idx] = __expf(sp * Ah);
}

// ---------------------------------------------------------------------------
// Selective scan: one block (32 lanes) per (b, head, p); lane owns 2 states (n)
// ---------------------------------------------------------------------------
__global__ __launch_bounds__(32) void scan_kernel(const float* __restrict__ xBC,
                                                  const float* __restrict__ dt,
                                                  const float* __restrict__ dexp,
                                                  float* __restrict__ yscan, int S) {
    int bid = blockIdx.x;
    int p = bid & 127, h = (bid >> 7) & 7, b = bid >> 10;
    int l = threadIdx.x;
    float s0 = 0.f, s1 = 0.f;
    int n0 = 2 * l;
    for (int t = 0; t < S; t++) {
        int row = b * S + t;
        float dtv = dt[row * 8 + h];
        float dec = dexp[row * 8 + h];
        float xv  = xBC[(size_t)row * 1152 + h * 128 + p];
        float2 Bv = *(const float2*)(xBC + (size_t)row * 1152 + 1024 + n0);
        float2 Cv = *(const float2*)(xBC + (size_t)row * 1152 + 1088 + n0);
        float c = dtv * xv;
        s0 = s0 * dec + c * Bv.x;
        s1 = s1 * dec + c * Bv.y;
        float part = s0 * Cv.x + s1 * Cv.y;
        #pragma unroll
        for (int o = 16; o > 0; o >>= 1) part += __shfl_xor(part, o, 32);
        if (l == 0) yscan[(size_t)row * 1024 + h * 128 + p] = part;
    }
}

// ---------------------------------------------------------------------------
// y = (yscan + D*xh) * silu(z); RMSNorm over 1024; * rms_w; bf16 out
// ---------------------------------------------------------------------------
__global__ __launch_bounds__(256) void gate_rms_kernel(const float* __restrict__ yscan,
                                                       const float* __restrict__ xBC,
                                                       const float* __restrict__ zx,
                                                       const float* __restrict__ Ds,
                                                       const float* __restrict__ rw,
                                                       bf16* __restrict__ out) {
    int row = blockIdx.x, tid = threadIdx.x;
    __shared__ float red[256];
    float g[4]; float ss = 0.f;
    #pragma unroll
    for (int i = 0; i < 4; i++) {
        int c = tid + i * 256;
        float xh = xBC[(size_t)row * 1152 + c];
        float yv = yscan[(size_t)row * 1024 + c] + Ds[c >> 7] * xh;
        float z  = zx[(size_t)row * 2184 + c];
        float gv = yv * (z / (1.f + __expf(-z)));
        g[i] = gv; ss += gv * gv;
    }
    red[tid] = ss; __syncthreads();
    for (int o = 128; o > 0; o >>= 1) { if (tid < o) red[tid] += red[tid + o]; __syncthreads(); }
    float scale = rsqrtf(red[0] / 1024.f + 1e-5f);
    #pragma unroll
    for (int i = 0; i < 4; i++) {
        int c = tid + i * 256;
        out[(size_t)row * 1024 + c] = f2bf(g[i] * scale * rw[c]);
    }
}

// ---------------------------------------------------------------------------
// K transpose: [B*M, 512] -> [B,H,64,M] bf16 for coalesced WMMA B-operand loads
// ---------------------------------------------------------------------------
__global__ void ktrans_kernel(const bf16* __restrict__ k, bf16* __restrict__ kT, int total) {
    int idx = blockIdx.x * blockDim.x + threadIdx.x;
    if (idx >= total) return;
    int col = idx & 511, row = idx >> 9;
    int b = row >> 10, m = row & 1023;
    int h = col >> 6, d = col & 63;
    kT[(size_t)((b * 8 + h) * 64 + d) * 1024 + m] = k[idx];
}

// ---------------------------------------------------------------------------
// Cross-attention: one block per (b, head, 16-query tile). WMMA QK^T -> LDS
// bf16 scores -> wave softmax -> WMMA PV with K split across waves.
// ---------------------------------------------------------------------------
__global__ __launch_bounds__(256) void attn_kernel(const bf16* __restrict__ Q,
                                                   const bf16* __restrict__ KT,
                                                   const bf16* __restrict__ V,
                                                   bf16* __restrict__ O,
                                                   int S, int Mk) {
    const int H = 8, DH = 64, DM = 512;
    int blk = blockIdx.x;
    int qt = blk & 127, h = (blk >> 7) & 7, b = blk >> 10;
    int q0 = qt * 16;
    int tid = threadIdx.x, lane = tid & 31, w = tid >> 5;
    __shared__ __attribute__((aligned(16))) bf16 sc[16 * 1032];
    __shared__ float pp[4 * 8 * 32];

    int base0 = (lane < 16) ? 0 : 8;
    int m = lane & 15;

    // ---- Q @ K^T (each wave: 128 keys) ----
    v8f acc[8];
    #pragma unroll
    for (int nt = 0; nt < 8; nt++) acc[nt] = (v8f){0.f,0.f,0.f,0.f,0.f,0.f,0.f,0.f};
    const bf16* qrow = Q + (size_t)(b * S + q0 + m) * DM + h * DH;
    #pragma unroll
    for (int kk = 0; kk < 2; kk++) {
        BFrag a;
        a.f4[0] = *(const float4*)(qrow + kk * 32 + base0);
        a.f4[1] = *(const float4*)(qrow + kk * 32 + base0 + 16);
        #pragma unroll
        for (int nt = 0; nt < 8; nt++) {
            int kb = w * 128 + nt * 16;
            BFrag bb;
            const bf16* kp = KT + (size_t)((b * H + h) * DH + kk * 32) * Mk + kb + m;
            #pragma unroll
            for (int s = 0; s < 8; s++) {
                bb.h[s]     = kp[(size_t)(base0 + s) * Mk];
                bb.h[s + 8] = kp[(size_t)(base0 + 16 + s) * Mk];
            }
            acc[nt] = wmma_bf16(a.v, bb.v, acc[nt]);
        }
    }
    #pragma unroll
    for (int nt = 0; nt < 8; nt++)
        #pragma unroll
        for (int r = 0; r < 8; r++) {
            int row = (lane < 16) ? r : r + 8;
            int col = w * 128 + nt * 16 + (lane & 15);
            sc[row * 1032 + col] = f2bf(acc[nt][r] * 0.125f);
        }
    __syncthreads();

    // ---- softmax: wave w handles rows 2w, 2w+1 ----
    #pragma unroll
    for (int rr = 0; rr < 2; rr++) {
        int row = w * 2 + rr;
        float vals[32];
        float mx = -1e30f;
        #pragma unroll
        for (int i = 0; i < 32; i++) {
            float xv = bf2f(sc[row * 1032 + lane + 32 * i]);
            vals[i] = xv; mx = fmaxf(mx, xv);
        }
        #pragma unroll
        for (int o = 16; o > 0; o >>= 1) mx = fmaxf(mx, __shfl_xor(mx, o, 32));
        float sum = 0.f;
        #pragma unroll
        for (int i = 0; i < 32; i++) { vals[i] = __expf(vals[i] - mx); sum += vals[i]; }
        #pragma unroll
        for (int o = 16; o > 0; o >>= 1) sum += __shfl_xor(sum, o, 32);
        float inv = 1.f / sum;
        #pragma unroll
        for (int i = 0; i < 32; i++) sc[row * 1032 + lane + 32 * i] = f2bf(vals[i] * inv);
    }
    __syncthreads();

    // ---- P @ V (wave w: ntile = w&3, k-half = w>>2) ----
    int nt2 = w & 3, kh = w >> 2;
    v8f oacc = (v8f){0.f,0.f,0.f,0.f,0.f,0.f,0.f,0.f};
    for (int kc = kh * 16; kc < kh * 16 + 16; kc++) {
        BFrag a;
        const bf16* prow = sc + m * 1032 + kc * 32;
        a.f4[0] = *(const float4*)(prow + base0);
        a.f4[1] = *(const float4*)(prow + base0 + 16);
        BFrag bb;
        const bf16* vp = V + (size_t)(b * Mk + kc * 32) * DM + h * DH + nt2 * 16 + m;
        #pragma unroll
        for (int s = 0; s < 8; s++) {
            bb.h[s]     = vp[(size_t)(base0 + s) * DM];
            bb.h[s + 8] = vp[(size_t)(base0 + 16 + s) * DM];
        }
        oacc = wmma_bf16(a.v, bb.v, oacc);
    }
    if (w >= 4) {
        #pragma unroll
        for (int r = 0; r < 8; r++) pp[(nt2 * 8 + r) * 32 + lane] = oacc[r];
    }
    __syncthreads();
    if (w < 4) {
        #pragma unroll
        for (int r = 0; r < 8; r++) {
            float vv = oacc[r] + pp[(nt2 * 8 + r) * 32 + lane];
            int gm = b * S + q0 + ((lane < 16) ? r : r + 8);
            int gn = h * DH + nt2 * 16 + (lane & 15);
            O[(size_t)gm * DM + gn] = f2bf(vv);
        }
    }
}

// ---------------------------------------------------------------------------
extern "C" void kernel_launch(void* const* d_in, const int* in_sizes, int n_in,
                              void* d_out, int out_size, void* d_ws, size_t ws_size,
                              hipStream_t stream) {
    const float* x       = (const float*)d_in[0];
    const float* memory  = (const float*)d_in[1];
    const float* pad     = (const float*)d_in[2];
    const float* ln1_g   = (const float*)d_in[3];
    const float* ln1_b   = (const float*)d_in[4];
    const float* ln2_g   = (const float*)d_in[5];
    const float* ln2_b   = (const float*)d_in[6];
    const float* W_in    = (const float*)d_in[7];
    const float* conv_w  = (const float*)d_in[8];
    const float* conv_b  = (const float*)d_in[9];
    const float* dt_bias = (const float*)d_in[10];
    const float* A_log   = (const float*)d_in[11];
    const float* D_skip  = (const float*)d_in[12];
    const float* rms_w   = (const float*)d_in[13];
    const float* W_out   = (const float*)d_in[14];
    const float* Wqkv    = (const float*)d_in[15];
    const float* bqkv    = (const float*)d_in[16];
    const float* Wo      = (const float*)d_in[17];
    const float* bo      = (const float*)d_in[18];

    const int Bn = 2, S = 2048, Mm = 1024;
    const int rows = Bn * S;     // 4096
    const int mrows = Bn * Mm;   // 2048

    char* ws = (char*)d_ws;
    size_t off = 0;
    auto alloc = [&](size_t bytes) -> char* {
        char* p = ws + off;
        off += (bytes + 255) & ~(size_t)255;
        return p;
    };
    bf16* wInB  = (bf16*)alloc((size_t)2184 * 512 * 2);
    bf16* wOutB = (bf16*)alloc((size_t)512 * 1024 * 2);
    bf16* wQkvB = (bf16*)alloc((size_t)1536 * 512 * 2);
    bf16* wOB   = (bf16*)alloc((size_t)512 * 512 * 2);
    bf16* memB  = (bf16*)alloc((size_t)mrows * 512 * 2);
    bf16* xnB   = (bf16*)alloc((size_t)rows * 512 * 2);
    float* zx   = (float*)alloc((size_t)rows * 2184 * 4);
    float* xBC  = (float*)alloc((size_t)rows * 1152 * 4);
    float* dtb  = (float*)alloc((size_t)rows * 8 * 4);
    float* dexp = (float*)alloc((size_t)rows * 8 * 4);
    float* ysc  = (float*)alloc((size_t)rows * 1024 * 4);
    bf16* ybf   = (bf16*)alloc((size_t)rows * 1024 * 2);
    float* x1   = (float*)alloc((size_t)rows * 512 * 4);
    bf16* hB    = (bf16*)alloc((size_t)rows * 512 * 2);
    bf16* qB    = (bf16*)alloc((size_t)rows * 512 * 2);
    bf16* kB    = (bf16*)alloc((size_t)mrows * 512 * 2);
    bf16* vB    = (bf16*)alloc((size_t)mrows * 512 * 2);
    bf16* kT    = (bf16*)alloc((size_t)mrows * 512 * 2);
    bf16* attnB = (bf16*)alloc((size_t)rows * 512 * 2);

    auto cdiv = [](int a, int b) { return (a + b - 1) / b; };

    // weight / memory converts
    cvt_bf16_kernel<<<cdiv(2184 * 512, 256), 256, 0, stream>>>(W_in, wInB, 2184 * 512);
    cvt_bf16_kernel<<<cdiv(512 * 1024, 256), 256, 0, stream>>>(W_out, wOutB, 512 * 1024);
    cvt_bf16_kernel<<<cdiv(1536 * 512, 256), 256, 0, stream>>>(Wqkv, wQkvB, 1536 * 512);
    cvt_bf16_kernel<<<cdiv(512 * 512, 256), 256, 0, stream>>>(Wo, wOB, 512 * 512);
    cvt_bf16_kernel<<<cdiv(mrows * 512, 256), 256, 0, stream>>>(memory, memB, mrows * 512);

    // LN1 + mask -> bf16
    ln_kernel<<<rows, 256, 0, stream>>>(x, ln1_g, ln1_b, pad, xnB);

    // zxbcdt = xn @ W_in^T
    gemm_bf16_kernel<<<dim3(cdiv(2184, 128), rows / 64), 256, 0, stream>>>(
        xnB, wInB, nullptr, nullptr, zx, nullptr, rows, 2184, 512);

    // conv + silu, dt/decay precompute
    conv_silu_kernel<<<cdiv(rows * 1152, 256), 256, 0, stream>>>(zx, conv_w, conv_b, xBC, S, rows * 1152);
    dt_kernel<<<cdiv(rows * 8, 256), 256, 0, stream>>>(zx, dt_bias, A_log, dtb, dexp, rows * 8);

    // selective scan
    scan_kernel<<<Bn * 8 * 128, 32, 0, stream>>>(xBC, dtb, dexp, ysc, S);

    // gate + RMSNorm -> bf16
    gate_rms_kernel<<<rows, 256, 0, stream>>>(ysc, xBC, zx, D_skip, rms_w, ybf);

    // x1 = x + y @ W_out^T
    gemm_bf16_kernel<<<dim3(cdiv(512, 128), rows / 64), 256, 0, stream>>>(
        ybf, wOutB, nullptr, x, x1, nullptr, rows, 512, 1024);

    // LN2 -> bf16
    ln_kernel<<<rows, 256, 0, stream>>>(x1, ln2_g, ln2_b, nullptr, hB);

    // Q / K / V projections (bf16 outputs)
    gemm_bf16_kernel<<<dim3(cdiv(512, 128), rows / 64), 256, 0, stream>>>(
        hB, wQkvB, bqkv, nullptr, nullptr, qB, rows, 512, 512);
    gemm_bf16_kernel<<<dim3(cdiv(512, 128), mrows / 64), 256, 0, stream>>>(
        memB, wQkvB + (size_t)512 * 512, bqkv + 512, nullptr, nullptr, kB, mrows, 512, 512);
    gemm_bf16_kernel<<<dim3(cdiv(512, 128), mrows / 64), 256, 0, stream>>>(
        memB, wQkvB + (size_t)1024 * 512, bqkv + 1024, nullptr, nullptr, vB, mrows, 512, 512);

    // K transpose for WMMA B operand
    ktrans_kernel<<<cdiv(mrows * 512, 256), 256, 0, stream>>>(kB, kT, mrows * 512);

    // attention
    attn_kernel<<<Bn * 8 * (S / 16), 256, 0, stream>>>(qB, kT, vB, attnB, S, Mm);

    // out = x1 + attn @ Wo^T + bo
    gemm_bf16_kernel<<<dim3(cdiv(512, 128), rows / 64), 256, 0, stream>>>(
        attnB, wOB, bo, x1, (float*)d_out, nullptr, rows, 512, 512);
}
